// MultiEdgeGraphBlock_11974368821918
// MI455X (gfx1250) — matrix operand order repaired and checked by reading, not compiled, not run on hardware
//
#include <hip/hip_runtime.h>
#include <hip/hip_bf16.h>

typedef __attribute__((ext_vector_type(16))) __bf16 v16bf;
typedef __attribute__((ext_vector_type(8)))  __bf16 v8bf;
typedef __attribute__((ext_vector_type(8)))  float  v8f;

namespace cfg {
constexpr int B = 2, N = 16384, F = 256, DEG = 16, E = 5, HD = 64;
constexpr int M  = B * N;     // 32768 GEMM rows
constexpr int F3 = 3 * F;     // 768
constexpr int GCH = 64;       // column-reduce chunks
}

// ---------------- elementwise f32 -> bf16 ----------------
__global__ void megb_f32_to_bf16(const float* __restrict__ X, __bf16* __restrict__ Y, int n) {
    int i = blockIdx.x * blockDim.x + threadIdx.x;
    if (i < n) Y[i] = (__bf16)X[i];
}

// ------- pack [K x Nc] row-major f32 weight into WMMA B-fragment bf16 -------
// layout: ((ntile*KT + kt)*32 + lane)*16 + e ; h=lane>>4: K=kt*32+8h+e (e<8), kt*32+16+8h+(e-8) (e>=8)
//                                             n=ntile*16+(lane&15)
__global__ void megb_pack_b(const float* __restrict__ W, __bf16* __restrict__ P, int K, int Nc) {
    int t = blockIdx.x * blockDim.x + threadIdx.x;
    if (t >= K * Nc) return;
    int e    = t & 15;
    int lane = (t >> 4) & 31;
    int q    = t >> 9;
    int KT   = K >> 5;
    int kt    = q % KT;
    int ntile = q / KT;
    int n = ntile * 16 + (lane & 15);
    int k = kt * 32 + ((lane & 16) ? 8 : 0) + (e & 7) + ((e & 8) ? 16 : 0);
    P[t] = (__bf16)W[(size_t)k * Nc + n];
}

// ------- bf16 WMMA GEMM: C[M,Nc] = A[M,K] * Bpacked[K,Nc] (+bias,+epilogue) -------
// Wave tile: 32 rows x 64 cols = 2 m-tiles x 4 n-tiles (B frags reused across both m-tiles).
// EPI 0: f32 store (+bias). EPI 1: bf16 relu(+bias). EPI 2: f32 (+bias +residual).
template <int EPI>
__global__ __launch_bounds__(256) void megb_gemm_bf16(
    const __bf16* __restrict__ A, const __bf16* __restrict__ Bp,
    const float* __restrict__ bias, const float* __restrict__ resid,
    void* __restrict__ Cout, int K, int Nc)
{
    const int lane   = threadIdx.x & 31;
    const int wave   = threadIdx.x >> 5;
    const int mTile0 = (blockIdx.x * 8 + wave) * 2;   // two consecutive 16-row tiles
    const int nBase  = blockIdx.y * 64;               // 4 x 16-col tiles per wave
    const int KT     = K >> 5;
    const int row0   = mTile0 * 16 + (lane & 15);
    const int aSub   = (lane & 16) ? 8 : 0;           // A frag: lanes>=16 hold K+8 / K+24 runs
    const __bf16* aRow0 = A + (size_t)row0 * K;
    const __bf16* aRow1 = aRow0 + (size_t)16 * K;

    v8f acc00 = {}, acc01 = {}, acc02 = {}, acc03 = {};
    v8f acc10 = {}, acc11 = {}, acc12 = {}, acc13 = {};
    for (int kt = 0; kt < KT; ++kt) {
        v8bf a0lo = *(const v8bf*)(aRow0 + kt * 32 + aSub);       // K = base+0..7 (or 8..15)
        v8bf a0hi = *(const v8bf*)(aRow0 + kt * 32 + aSub + 16);  // K = base+16..23 (or 24..31)
        v8bf a1lo = *(const v8bf*)(aRow1 + kt * 32 + aSub);
        v8bf a1hi = *(const v8bf*)(aRow1 + kt * 32 + aSub + 16);
        v16bf a0, a1;
#pragma unroll
        for (int i = 0; i < 8; ++i) {
            a0[i] = a0lo[i]; a0[i + 8] = a0hi[i];
            a1[i] = a1lo[i]; a1[i + 8] = a1hi[i];
        }

        const __bf16* bb = Bp + ((size_t)((nBase >> 4) * KT + kt) * 32 + lane) * 16;
        v16bf b0 = *(const v16bf*)(bb);
        v16bf b1 = *(const v16bf*)(bb + (size_t)KT * 512);
        v16bf b2 = *(const v16bf*)(bb + (size_t)KT * 1024);
        v16bf b3 = *(const v16bf*)(bb + (size_t)KT * 1536);

        acc00 = __builtin_amdgcn_wmma_f32_16x16x32_bf16(false, a0, false, b0, (short)0, acc00, false, false);
        acc01 = __builtin_amdgcn_wmma_f32_16x16x32_bf16(false, a0, false, b1, (short)0, acc01, false, false);
        acc02 = __builtin_amdgcn_wmma_f32_16x16x32_bf16(false, a0, false, b2, (short)0, acc02, false, false);
        acc03 = __builtin_amdgcn_wmma_f32_16x16x32_bf16(false, a0, false, b3, (short)0, acc03, false, false);
        acc10 = __builtin_amdgcn_wmma_f32_16x16x32_bf16(false, a1, false, b0, (short)0, acc10, false, false);
        acc11 = __builtin_amdgcn_wmma_f32_16x16x32_bf16(false, a1, false, b1, (short)0, acc11, false, false);
        acc12 = __builtin_amdgcn_wmma_f32_16x16x32_bf16(false, a1, false, b2, (short)0, acc12, false, false);
        acc13 = __builtin_amdgcn_wmma_f32_16x16x32_bf16(false, a1, false, b3, (short)0, acc13, false, false);
    }

    const int rowHi = (lane & 16) ? 8 : 0;  // C/D frag: lanes>=16 hold rows r+8
#pragma unroll
    for (int mi = 0; mi < 2; ++mi) {
#pragma unroll
        for (int j = 0; j < 4; ++j) {
            v8f acc;
            if (mi == 0) acc = (j == 0) ? acc00 : (j == 1) ? acc01 : (j == 2) ? acc02 : acc03;
            else         acc = (j == 0) ? acc10 : (j == 1) ? acc11 : (j == 2) ? acc12 : acc13;
            int col = nBase + j * 16 + (lane & 15);
            float bv = bias ? bias[col] : 0.f;
#pragma unroll
            for (int r = 0; r < 8; ++r) {
                size_t o = (size_t)((mTile0 + mi) * 16 + r + rowHi) * Nc + col;
                float v = acc[r] + bv;
                if (EPI == 1) {
                    ((__bf16*)Cout)[o] = (__bf16)fmaxf(v, 0.f);
                } else if (EPI == 2) {
                    ((float*)Cout)[o] = v + resid[o];
                } else {
                    ((float*)Cout)[o] = v;
                }
            }
        }
    }
}

// -------- fused gather + masked softmax + weighted-P aggregation (one wave per node) --------
__global__ __launch_bounds__(256) void megb_edge_attn(
    const float* __restrict__ Q, const float* __restrict__ Kv, const float* __restrict__ P,
    const int* __restrict__ eidx, const unsigned char* __restrict__ emask,
    const float* __restrict__ bproj, float* __restrict__ agg, int initFlag)
{
    const int lane = threadIdx.x & 31;
    const int wave = threadIdx.x >> 5;
    const int n = blockIdx.x * 8 + wave;
    const int b = blockIdx.y;
    const size_t row = (size_t)b * cfg::N + n;

    float q0 = Q[row * 64 + lane];
    float q1 = Q[row * 64 + 32 + lane];

    int   ids[16];
    float lg[16];
#pragma unroll
    for (int d = 0; d < 16; ++d) {
        int id = eidx[n * 16 + d]; id = id < 0 ? 0 : id;
        ids[d] = id;
        const float* kp = Kv + ((size_t)b * cfg::N + id) * 64;
        float p = q0 * kp[lane] + q1 * kp[32 + lane];
#pragma unroll
        for (int o = 16; o > 0; o >>= 1) p += __shfl_xor(p, o, 32);
        lg[d] = p * 0.125f;  // 1/sqrt(64)
    }
    bool mk[16]; float mx = -1e30f;
#pragma unroll
    for (int d = 0; d < 16; ++d) {
        mk[d] = emask[n * 16 + d] != 0;
        if (!mk[d]) lg[d] = -1e9f;
        mx = fmaxf(mx, lg[d]);
    }
    float w[16]; float den = 0.f;
#pragma unroll
    for (int d = 0; d < 16; ++d) { float ev = mk[d] ? __expf(lg[d] - mx) : 0.f; w[d] = ev; den += ev; }
    float inv = den > 0.f ? 1.f / den : 0.f;

    float acc[8];
#pragma unroll
    for (int j = 0; j < 8; ++j) acc[j] = 0.f;
#pragma unroll
    for (int d = 0; d < 16; ++d) {
        float wd = w[d] * inv;
        if (wd != 0.f) {
            const float* pp = P + ((size_t)b * cfg::N + ids[d]) * 256;
#pragma unroll
            for (int j = 0; j < 8; ++j) acc[j] += wd * pp[lane + 32 * j];
        }
    }
    float* ap = agg + row * 256;
#pragma unroll
    for (int j = 0; j < 8; ++j) {
        float v = acc[j] + bproj[lane + 32 * j];
        if (initFlag) ap[lane + 32 * j] = v; else ap[lane + 32 * j] += v;
    }
}

// -------- fused concat([h,agg,g]) + LayerNorm(768), bf16 output (one wave per node) --------
__global__ __launch_bounds__(256) void megb_ln_concat(
    const float* __restrict__ h, const float* __restrict__ agg, const float* __restrict__ g,
    const float* __restrict__ lns, const float* __restrict__ lnb, __bf16* __restrict__ xcat)
{
    const int lane = threadIdx.x & 31;
    const int wave = threadIdx.x >> 5;
    const int n = blockIdx.x * 8 + wave;
    const int b = blockIdx.y;
    const size_t row = (size_t)b * cfg::N + n;

    float xv[24]; float s = 0.f, s2 = 0.f;
#pragma unroll
    for (int j = 0; j < 24; ++j) {
        int t = j * 32 + lane;
        float v;
        if (j < 8)       v = h[row * 256 + t];
        else if (j < 16) v = agg[row * 256 + (t - 256)];
        else             v = g[b * 256 + (t - 512)];
        xv[j] = v; s += v; s2 += v * v;
    }
#pragma unroll
    for (int o = 16; o > 0; o >>= 1) { s += __shfl_xor(s, o, 32); s2 += __shfl_xor(s2, o, 32); }
    float mu   = s * (1.f / 768.f);
    float var  = s2 * (1.f / 768.f) - mu * mu;
    float rstd = rsqrtf(var + 1e-6f);
#pragma unroll
    for (int j = 0; j < 24; ++j) {
        int t = j * 32 + lane;
        float v = (xv[j] - mu) * rstd * lns[t] + lnb[t];
        xcat[row * 768 + t] = (__bf16)v;
    }
}

// ---------------- deterministic two-stage column mean over N ----------------
__global__ __launch_bounds__(256) void megb_col_sum_partial(
    const float* __restrict__ X, float* __restrict__ part, int rowsPer)
{
    int f = threadIdx.x, ch = blockIdx.x, b = blockIdx.y;
    size_t base = ((size_t)b * cfg::N + (size_t)ch * rowsPer) * 256 + f;
    float s = 0.f;
    for (int r = 0; r < rowsPer; ++r) s += X[base + (size_t)r * 256];
    part[((size_t)b * gridDim.x + ch) * 256 + f] = s;
}

__global__ void megb_col_sum_final(const float* __restrict__ part, float* __restrict__ out,
                                   int chunks, float scale)
{
    int f = threadIdx.x, b = blockIdx.x;
    float s = 0.f;
    for (int c = 0; c < chunks; ++c) s += part[((size_t)b * chunks + c) * 256 + f];
    out[b * 256 + f] = s * scale;
}

// ---------------- tiny SE MLP: relu(sep@Wse1+b) -> sigmoid(@Wse2+b) ----------------
__global__ __launch_bounds__(256) void megb_se(
    const float* __restrict__ sep, const float* __restrict__ Wse1, const float* __restrict__ bse1,
    const float* __restrict__ Wse2, const float* __restrict__ bse2, float* __restrict__ se)
{
    __shared__ float sb[256];
    __shared__ float t1[64];
    for (int b = 0; b < cfg::B; ++b) {
        sb[threadIdx.x] = sep[b * 256 + threadIdx.x];
        __syncthreads();
        if (threadIdx.x < 64) {
            float s = bse1[threadIdx.x];
            for (int f = 0; f < 256; ++f) s += sb[f] * Wse1[f * 64 + threadIdx.x];
            t1[threadIdx.x] = fmaxf(s, 0.f);
        }
        __syncthreads();
        {
            float s = bse2[threadIdx.x];
            for (int j = 0; j < 64; ++j) s += t1[j] * Wse2[j * 256 + threadIdx.x];
            se[b * 256 + threadIdx.x] = 1.f / (1.f + __expf(-s));
        }
        __syncthreads();
    }
}

// ---------------- out = out_pre * se (broadcast over n) ----------------
__global__ void megb_scale_out(const float* __restrict__ pre, const float* __restrict__ se,
                               float* __restrict__ out)
{
    size_t i = (size_t)blockIdx.x * blockDim.x + threadIdx.x;
    int f = (int)(i & 255);
    int b = (int)(i >> 22);  // N*F = 2^22
    out[i] = pre[i] * se[b * 256 + f];
}

extern "C" void kernel_launch(void* const* d_in, const int* in_sizes, int n_in,
                              void* d_out, int out_size, void* d_ws, size_t ws_size,
                              hipStream_t stream)
{
    using namespace cfg;
    const float* h     = (const float*)d_in[0];
    const float* Wq    = (const float*)d_in[1];
    const float* bq    = (const float*)d_in[2];
    const float* Wk    = (const float*)d_in[3];
    const float* bk    = (const float*)d_in[4];
    const float* Wproj = (const float*)d_in[5];
    const float* bproj = (const float*)d_in[6];
    const float* lns   = (const float*)d_in[7];
    const float* lnb   = (const float*)d_in[8];
    const float* W1    = (const float*)d_in[9];
    const float* b1    = (const float*)d_in[10];
    const float* W2    = (const float*)d_in[11];
    const float* b2    = (const float*)d_in[12];
    const float* Wse1  = (const float*)d_in[13];
    const float* bse1  = (const float*)d_in[14];
    const float* Wse2  = (const float*)d_in[15];
    const float* bse2  = (const float*)d_in[16];
    const int*   eidx  = (const int*)d_in[17];
    const unsigned char* emask = (const unsigned char*)d_in[18];
    float* out = (float*)d_out;
    (void)in_sizes; (void)n_in; (void)out_size; (void)ws_size;

    char* ws = (char*)d_ws;
    size_t cur = 0;
    auto alloc = [&](size_t bytes) -> void* {
        void* p = ws + cur; cur += (bytes + 255) & ~(size_t)255; return p;
    };

    __bf16* hbf  = (__bf16*)alloc((size_t)M * F * 2);
    float*  agg  = (float*) alloc((size_t)M * F * 4);               // reused later as out_pre
    char*   qkp  = (char*)  alloc((size_t)M * (64 + 64 + 256) * 4); // Q|K|P ; reused as xcat
    float*  Qb   = (float*)qkp;
    float*  Kb   = (float*)(qkp + (size_t)M * 64 * 4);
    float*  Pb   = (float*)(qkp + (size_t)M * 128 * 4);
    __bf16* xcat = (__bf16*)qkp;                                    // M*768 bf16, same bytes
    __bf16* y1bf = (__bf16*)alloc((size_t)M * F * 2);
    float*  gpart = (float*)alloc((size_t)B * GCH * F * 4);
    float*  gvec  = (float*)alloc((size_t)B * F * 4);
    float*  spart = (float*)alloc((size_t)B * GCH * F * 4);
    float*  sevec = (float*)alloc((size_t)B * F * 4);
    __bf16* WqP  = (__bf16*)alloc((size_t)E * F * HD * 2);
    __bf16* WkP  = (__bf16*)alloc((size_t)E * F * HD * 2);
    __bf16* WpjP = (__bf16*)alloc((size_t)E * F * F * 2);
    __bf16* W1P  = (__bf16*)alloc((size_t)F3 * F * 2);
    __bf16* W2P  = (__bf16*)alloc((size_t)F * F * 2);

    // h -> bf16
    { int n = M * F; megb_f32_to_bf16<<<(n + 255) / 256, 256, 0, stream>>>(h, hbf, n); }

    // pack all weight matrices into WMMA B-fragment layout (bf16)
    for (int e = 0; e < E; ++e) {
        megb_pack_b<<<(F * HD + 255) / 256, 256, 0, stream>>>(Wq + (size_t)e * F * HD, WqP + (size_t)e * F * HD, F, HD);
        megb_pack_b<<<(F * HD + 255) / 256, 256, 0, stream>>>(Wk + (size_t)e * F * HD, WkP + (size_t)e * F * HD, F, HD);
        megb_pack_b<<<(F * F + 255) / 256, 256, 0, stream>>>(Wproj + (size_t)e * F * F, WpjP + (size_t)e * F * F, F, F);
    }
    megb_pack_b<<<(F3 * F + 255) / 256, 256, 0, stream>>>(W1, W1P, F3, F);
    megb_pack_b<<<(F * F + 255) / 256, 256, 0, stream>>>(W2, W2P, F, F);

    // g = mean over nodes of h
    megb_col_sum_partial<<<dim3(GCH, B), 256, 0, stream>>>(h, gpart, N / GCH);
    megb_col_sum_final<<<B, 256, 0, stream>>>(gpart, gvec, GCH, 1.f / (float)N);

    // GEMM grids: 8 waves/block, each wave covers 32 rows -> 256 rows per block
    dim3 gGemm64(M / 256, 1), gGemm256(M / 256, 4);
    for (int e = 0; e < E; ++e) {
        megb_gemm_bf16<0><<<gGemm64, 256, 0, stream>>>(hbf, WqP + (size_t)e * F * HD, bq + (size_t)e * HD, nullptr, Qb, F, HD);
        megb_gemm_bf16<0><<<gGemm64, 256, 0, stream>>>(hbf, WkP + (size_t)e * F * HD, bk + (size_t)e * HD, nullptr, Kb, F, HD);
        megb_gemm_bf16<0><<<gGemm256, 256, 0, stream>>>(hbf, WpjP + (size_t)e * F * F, nullptr, nullptr, Pb, F, F);
        megb_edge_attn<<<dim3(N / 8, B), 256, 0, stream>>>(Qb, Kb, Pb,
                eidx + (size_t)e * N * DEG, emask + (size_t)e * N * DEG,
                bproj + (size_t)e * F, agg, e == 0 ? 1 : 0);
    }

    // concat + LN -> bf16  (overwrites Q|K|P region, which is dead now)
    megb_ln_concat<<<dim3(N / 8, B), 256, 0, stream>>>(h, agg, gvec, lns, lnb, xcat);

    // MLP: y1 = relu(xcat@W1 + b1) [bf16], out_pre = h + y1@W2 + b2 [f32, reuses agg]
    megb_gemm_bf16<1><<<gGemm256, 256, 0, stream>>>(xcat, W1P, b1, nullptr, y1bf, F3, F);
    float* outpre = agg;
    megb_gemm_bf16<2><<<gGemm256, 256, 0, stream>>>(y1bf, W2P, b2, h, outpre, F, F);

    // SE gate
    megb_col_sum_partial<<<dim3(GCH, B), 256, 0, stream>>>(outpre, spart, N / GCH);
    megb_col_sum_final<<<B, 256, 0, stream>>>(spart, sevec, GCH, 1.f / (float)N);
    megb_se<<<1, 256, 0, stream>>>(sevec, Wse1, bse1, Wse2, bse2, sevec);

    // final output
    megb_scale_out<<<(unsigned)((size_t)M * F / 256), 256, 0, stream>>>(outpre, sevec, out);
}
